// ImageEncoding_79465484911184
// MI455X (gfx1250) — matrix operand order repaired and checked by reading, not compiled, not run on hardware
//
#include <hip/hip_runtime.h>

#define GDIM 7
#define NREG 49
#define NB   4
#define NQ   100
#define CDIM 256
#define HREG 32
#define SREG 1024          // 32*32 positions per region
#define IMW  224
#define QT   16            // queries per tile
#define NQT  7             // ceil(100/16)
#define SCH  128           // s-chunk staged per iteration
#define NCHK 8             // 1024/128
#define KV_STRIDE 264      // u16 per staged row (256 + 8 pad -> conflict-free frag reads)
#define Q_STRIDE  264
#define P_STRIDE  1032

// dynamic-LDS layout (byte offsets)
#define OFF_S     0                       // f32 S[16][1024]            65536 B
#define OFF_QHI   65536                   // u16 Qhi[16][264]            8448 B
#define OFF_QLO   73984                   // u16 Qlo[16][264]            8448 B
#define OFF_KVHI  82432                   // u16 K/V hi [128][264]      67584 B
#define OFF_KLO   150016                  // u16 K lo [128][264]; later P[16][1032]
#define OFF_RED   217600                  // f32 red[16][16]             1024 B
#define OFF_STAT  218624                  // f32 stat[32]                 128 B
#define SMEM_BYTES 218752

typedef __bf16 bf16_t;
typedef bf16_t v16bf __attribute__((ext_vector_type(16)));
typedef float  v8f   __attribute__((ext_vector_type(8)));

union Frag { unsigned u[8]; v16bf v; };

__device__ __forceinline__ unsigned short f32_to_bf16(float x) {
  unsigned u = __float_as_uint(x);
  return (unsigned short)((u + 0x7FFFu + ((u >> 16) & 1u)) >> 16);
}
__device__ __forceinline__ float bf16_to_f32(unsigned short h) {
  return __uint_as_float(((unsigned)h) << 16);
}
__device__ __forceinline__ void split_bf16(float x, unsigned short& hi, unsigned short& lo) {
  hi = f32_to_bf16(x);
  lo = f32_to_bf16(x - bf16_to_f32(hi));
}

// A-matrix 16x32 bf16 fragment, contiguous k order (same as B loaders): lane m=l16, half=lane>>4;
// VGPR i holds K = 2i + 16*half, +1 (packed lo/hi 16 bits)
__device__ __forceinline__ void load_a_frag(const unsigned* base32, int stride16,
                                            int m, int k0, int half, Frag& f) {
#pragma unroll
  for (int i = 0; i < 8; ++i) {
    int k = k0 + 2 * i + 16 * half;
    f.u[i] = base32[(m * stride16 + k) >> 1];
  }
}
// B-matrix 32x16 bf16, K runs along the contiguous (channel) direction of row `row`:
// lane col n=l16 -> staged row; VGPR v holds K = 2v + 16*half, +1
__device__ __forceinline__ void load_b_frag_rows(const unsigned* base32, int stride16,
                                                 int row, int k0, int half, Frag& f) {
#pragma unroll
  for (int v = 0; v < 8; ++v) {
    int k = k0 + 2 * v + 16 * half;
    f.u[v] = base32[(row * stride16 + k) >> 1];
  }
}
// B-matrix 32x16 bf16, K runs across staged rows (s), N is the channel column
__device__ __forceinline__ void load_b_frag_cols(const unsigned short* base16, int stride16,
                                                 int s0, int col, int half, Frag& f) {
#pragma unroll
  for (int v = 0; v < 8; ++v) {
    int k = s0 + 2 * v + 16 * half;
    unsigned lo = base16[k * stride16 + col];
    unsigned hi = base16[(k + 1) * stride16 + col];
    f.u[v] = lo | (hi << 16);
  }
}

__global__ __launch_bounds__(256) void attn_region_kernel(
    const float* __restrict__ images, const float* __restrict__ queries,
    float* __restrict__ out) {
  extern __shared__ char smem[];
  float*          sS    = (float*)(smem + OFF_S);
  unsigned short* sKVhi = (unsigned short*)(smem + OFF_KVHI);
  unsigned short* sP    = (unsigned short*)(smem + OFF_KLO);
  float*          sRed  = (float*)(smem + OFF_RED);
  float*          sStat = (float*)(smem + OFF_STAT);

  const int tid  = (int)threadIdx.x;
  const int lane = tid & 31;
  const int wv   = tid >> 5;          // 0..7
  const int half = (lane >> 4) & 1;
  const int l16  = lane & 15;

  const int bx = (int)blockIdx.x;
  const int qt = bx % NQT;
  const int rn = (bx / NQT) % NREG;
  const int b  = bx / (NQT * NREG);
  const int gi = rn / GDIM;
  const int gj = rn % GDIM;

  const float* imgb = images + (size_t)b * IMW * IMW * CDIM;

  // ---- stage Q tile as hi/lo bf16 split ----
  {
    unsigned* qhi32 = (unsigned*)(smem + OFF_QHI);
    unsigned* qlo32 = (unsigned*)(smem + OFF_QLO);
#pragma unroll
    for (int i = 0; i < 4; ++i) {
      int f4  = tid + i * 256;          // 0..1023 float4s
      int row = f4 >> 6;                // 0..15
      int c4  = (f4 & 63) * 4;          // 0..252
      int q   = qt * QT + row;
      float4 d = make_float4(0.f, 0.f, 0.f, 0.f);
      if (q < NQ) d = *(const float4*)(queries + (size_t)q * CDIM + c4);
      unsigned short h0,l0,h1,l1,h2,l2,h3,l3;
      split_bf16(d.x,h0,l0); split_bf16(d.y,h1,l1);
      split_bf16(d.z,h2,l2); split_bf16(d.w,h3,l3);
      int o = row * Q_STRIDE + c4;
      qhi32[o >> 1]       = (unsigned)h0 | ((unsigned)h1 << 16);
      qhi32[(o + 2) >> 1] = (unsigned)h2 | ((unsigned)h3 << 16);
      qlo32[o >> 1]       = (unsigned)l0 | ((unsigned)l1 << 16);
      qlo32[(o + 2) >> 1] = (unsigned)l2 | ((unsigned)l3 << 16);
    }
  }

  const unsigned* qhi32  = (const unsigned*)(smem + OFF_QHI);
  const unsigned* qlo32  = (const unsigned*)(smem + OFF_QLO);
  const unsigned* kvhi32 = (const unsigned*)(smem + OFF_KVHI);
  const unsigned* klo32  = (const unsigned*)(smem + OFF_KLO);

  // ---- phase 1: S = Q K^T with hi/lo bf16 split (fp32-grade scores) ----
  for (int ch = 0; ch < NCHK; ++ch) {
    __syncthreads();
    {
      unsigned* khiW = (unsigned*)(smem + OFF_KVHI);
      unsigned* kloW = (unsigned*)(smem + OFF_KLO);
#pragma unroll
      for (int i = 0; i < 32; ++i) {
        int f4 = tid + i * 256;          // 0..8191
        int sl = f4 >> 6;                // 0..127
        int c4 = (f4 & 63) * 4;
        int s  = ch * SCH + sl;
        int hh = s >> 5, ww = s & 31;
        const float* gp = imgb + ((size_t)(gi * HREG + hh) * IMW + (gj * HREG + ww)) * CDIM + c4;
        const float4 d = *(const float4*)gp;
        if (ch + 1 < NCHK) {
          int sn = (ch + 1) * SCH + sl;
          __builtin_prefetch(imgb + ((size_t)(gi*HREG + (sn>>5)) * IMW + (gj*HREG + (sn&31))) * CDIM + c4, 0, 1);
        }
        unsigned short h0,l0,h1,l1,h2,l2,h3,l3;
        split_bf16(d.x,h0,l0); split_bf16(d.y,h1,l1);
        split_bf16(d.z,h2,l2); split_bf16(d.w,h3,l3);
        int o = sl * KV_STRIDE + c4;
        khiW[o >> 1]       = (unsigned)h0 | ((unsigned)h1 << 16);
        khiW[(o + 2) >> 1] = (unsigned)h2 | ((unsigned)h3 << 16);
        kloW[o >> 1]       = (unsigned)l0 | ((unsigned)l1 << 16);
        kloW[(o + 2) >> 1] = (unsigned)l2 | ((unsigned)l3 << 16);
      }
    }
    __syncthreads();

    v8f cS = {};
    const int srow = wv * 16 + l16;      // this wave's s-tile, lane's B column row
#pragma unroll
    for (int kk = 0; kk < 8; ++kk) {
      const int k0 = kk * 32;
      Frag aH, aL, bH, bL;
      load_a_frag(qhi32, Q_STRIDE, l16, k0, half, aH);
      load_a_frag(qlo32, Q_STRIDE, l16, k0, half, aL);
      load_b_frag_rows(kvhi32, KV_STRIDE, srow, k0, half, bH);
      load_b_frag_rows(klo32,  KV_STRIDE, srow, k0, half, bL);
      cS = __builtin_amdgcn_wmma_f32_16x16x32_bf16(false, aH.v, false, bH.v, (short)0, cS, false, false);
      cS = __builtin_amdgcn_wmma_f32_16x16x32_bf16(false, aH.v, false, bL.v, (short)0, cS, false, false);
      cS = __builtin_amdgcn_wmma_f32_16x16x32_bf16(false, aL.v, false, bH.v, (short)0, cS, false, false);
    }
    const int scol = ch * SCH + wv * 16 + l16;
#pragma unroll
    for (int r = 0; r < 8; ++r)
      sS[(r + 8 * half) * SREG + scol] = cS[r];
  }
  __syncthreads();

  // ---- softmax over s (exact fp32, same max-subtract algorithm as jax) ----
  const int rowq = tid >> 4;   // 0..15
  const int sub  = tid & 15;   // 0..15
  {
    float mx = -3.0e38f;
    for (int s = sub; s < SREG; s += 16) mx = fmaxf(mx, sS[rowq * SREG + s]);
    sRed[rowq * 16 + sub] = mx;
  }
  __syncthreads();
  if (sub == 0) {
    float m = sRed[rowq * 16];
#pragma unroll
    for (int i = 1; i < 16; ++i) m = fmaxf(m, sRed[rowq * 16 + i]);
    sStat[rowq] = m;
  }
  __syncthreads();
  const float rmax = sStat[rowq];
  {
    float sum = 0.f;
    for (int s = sub; s < SREG; s += 16) {
      float e = __expf(sS[rowq * SREG + s] - rmax);
      sS[rowq * SREG + s] = e;         // cache exp for the scaling pass
      sum += e;
    }
    __syncthreads();
    sRed[rowq * 16 + sub] = sum;
  }
  __syncthreads();
  if (sub == 0) {
    float t = 0.f;
#pragma unroll
    for (int i = 0; i < 16; ++i) t += sRed[rowq * 16 + i];
    sStat[16 + rowq] = 1.0f / t;
  }
  __syncthreads();
  {
    const float rinv = sStat[16 + rowq];
    for (int s = sub; s < SREG; s += 16)
      sP[rowq * P_STRIDE + s] = f32_to_bf16(sS[rowq * SREG + s] * rinv);
  }

  // ---- phase 2: out = P V (bf16, f32 accum); each wave owns 32 channels ----
  v8f acc0 = {}, acc1 = {};
  const int c0 = wv * 32;
  const unsigned* p32 = (const unsigned*)sP;
  for (int ch = 0; ch < NCHK; ++ch) {
    __syncthreads();
    {
      unsigned* vhiW = (unsigned*)(smem + OFF_KVHI);
#pragma unroll
      for (int i = 0; i < 32; ++i) {
        int f4 = tid + i * 256;
        int sl = f4 >> 6;
        int c4 = (f4 & 63) * 4;
        int s  = ch * SCH + sl;
        int hh = s >> 5, ww = s & 31;
        const float4 d = *(const float4*)(imgb + ((size_t)(gi * HREG + hh) * IMW + (gj * HREG + ww)) * CDIM + c4);
        int o = sl * KV_STRIDE + c4;
        vhiW[o >> 1]       = (unsigned)f32_to_bf16(d.x) | ((unsigned)f32_to_bf16(d.y) << 16);
        vhiW[(o + 2) >> 1] = (unsigned)f32_to_bf16(d.z) | ((unsigned)f32_to_bf16(d.w) << 16);
      }
    }
    __syncthreads();
#pragma unroll
    for (int kk = 0; kk < 4; ++kk) {
      Frag aP, b0, b1;
      load_a_frag(p32, P_STRIDE, l16, ch * SCH + kk * 32, half, aP);
      load_b_frag_cols(sKVhi, KV_STRIDE, kk * 32, c0 + l16,      half, b0);
      load_b_frag_cols(sKVhi, KV_STRIDE, kk * 32, c0 + 16 + l16, half, b1);
      acc0 = __builtin_amdgcn_wmma_f32_16x16x32_bf16(false, aP.v, false, b0.v, (short)0, acc0, false, false);
      acc1 = __builtin_amdgcn_wmma_f32_16x16x32_bf16(false, aP.v, false, b1.v, (short)0, acc1, false, false);
    }
  }

  // ---- store out[b][rn][q][c] ----
  float* outb = out + ((size_t)b * NREG + rn) * NQ * CDIM;
#pragma unroll
  for (int r = 0; r < 8; ++r) {
    int m = r + 8 * half;
    int q = qt * QT + m;
    if (q < NQ) {
      outb[(size_t)q * CDIM + c0 + l16]      = acc0[r];
      outb[(size_t)q * CDIM + c0 + 16 + l16] = acc1[r];
    }
  }
}

__global__ void positions_kernel(float* __restrict__ pos) {
  int t = (int)threadIdx.x;
  if (t < NREG) {
    pos[2 * t]     = (float)(t / GDIM) * (1.0f / 7.0f);
    pos[2 * t + 1] = (float)(t % GDIM) * (1.0f / 7.0f);
  }
}

extern "C" void kernel_launch(void* const* d_in, const int* in_sizes, int n_in,
                              void* d_out, int out_size, void* d_ws, size_t ws_size,
                              hipStream_t stream) {
  (void)in_sizes; (void)n_in; (void)out_size; (void)d_ws; (void)ws_size;
  const float* images  = (const float*)d_in[0];
  const float* queries = (const float*)d_in[1];
  float* out = (float*)d_out;

  attn_region_kernel<<<dim3(NB * NREG * NQT), dim3(256), SMEM_BYTES, stream>>>(images, queries, out);
  positions_kernel<<<dim3(1), dim3(64), 0, stream>>>(out + (size_t)NB * NREG * NQ * CDIM);
}
